// EuclideanCodebook_43748536877160
// MI455X (gfx1250) — compile-verified, matrix operand order checked by reading, not run on hardware
//
#include <hip/hip_runtime.h>

// ---------------------------------------------------------------------------
// EuclideanCodebook forward for MI455X (gfx1250, wave32, WMMA).
// x: (65536, 128) f32, embed: (1024, 128) f32.
// out = [quantize (65536*128) f32][embed_ind (65536) f32 values of indices]
// argmax_k (x.e_k - 0.5*||e_k||^2) via bf16 WMMA 16x16x32 (f32 accumulate).
// Whole bf16 codebook staged per block into WGP LDS (320KB on CDNA5),
// B fragments ping-ponged in registers against the WMMA pipe; argmax kept
// as packed f32 max (column index in low 10 mantissa bits). A per-element
// empty-asm barrier prevents the optimizer from collapsing the WMMA result
// elements (observed miscompile in round 3).
// ---------------------------------------------------------------------------

typedef __attribute__((ext_vector_type(16))) __bf16 v16bf;
typedef __attribute__((ext_vector_type(8)))  __bf16 v8bf;
typedef __attribute__((ext_vector_type(8)))  float  v8f;

#define DDIM   128
#define KCODES 1024
#define NTILES (KCODES / 16)
#define LROW   136   // LDS row pitch in bf16 (272B) -> <=2-way bank conflicts
#define IDXMASK 0xFFFFFC00u   // clear low 10 bits of mantissa for index pack

static __device__ __forceinline__ __bf16 f2bf(float f) {
    unsigned u = __builtin_bit_cast(unsigned, f);
    unsigned r = u + 0x7FFFu + ((u >> 16) & 1u);   // RNE to bf16
    unsigned short h = (unsigned short)(r >> 16);
    return __builtin_bit_cast(__bf16, h);
}

// One block per code: convert embed row to bf16, nbias = -0.5*||e||^2.
__global__ __launch_bounds__(DDIM)
void vq_prep_kernel(const float* __restrict__ embed,
                    __bf16* __restrict__ ebf,
                    float* __restrict__ nbias) {
    __shared__ float red[DDIM];
    const int k = blockIdx.x;
    const int d = threadIdx.x;
    const float v = embed[(size_t)k * DDIM + d];
    ebf[(size_t)k * DDIM + d] = f2bf(v);
    red[d] = v * v;
    __syncthreads();
    #pragma unroll
    for (int s = DDIM / 2; s > 0; s >>= 1) {
        if (d < s) red[d] += red[d + s];
        __syncthreads();
    }
    if (d == 0) nbias[k] = -0.5f * red[0];
}

// 256 threads = 8 waves; each wave owns 16 rows of x; block covers 128 rows.
__global__ __launch_bounds__(256)
void vq_main_kernel(const float* __restrict__ x,
                    const float* __restrict__ embed,
                    const __bf16* __restrict__ ebf,
                    const float* __restrict__ nbias,
                    float* __restrict__ quant,
                    float* __restrict__ idxf) {
    __shared__ __bf16 semb[KCODES * LROW];   // 272 KB: whole bf16 codebook
    __shared__ float  sbias[KCODES];         // 4 KB (negated bias)

    const int tid = threadIdx.x;

    // ---- stage codebook (bf16) + bias into LDS, padded rows ---------------
    for (int i = tid; i < KCODES * 16; i += 256) {       // 16B granules
        const int n = i >> 4, g = i & 15;
        const uint4 v = *(const uint4*)(ebf + (size_t)n * DDIM + g * 8);
        *(uint4*)(&semb[n * LROW + g * 8]) = v;
    }
    for (int i = tid; i < KCODES; i += 256) sbias[i] = nbias[i];
    __syncthreads();

    const int wave = tid >> 5;
    const int lane = tid & 31;
    const int hs   = lane >> 4;      // half-wave: K +8 offset group
    const int lrow = lane & 15;      // row (A) / column (B) within 16-tile
    const int m0   = blockIdx.x * 128 + wave * 16;

    // ---- A fragments: 16 rows x 128 dims of x, bf16, resident in VGPRs ----
    // 16-bit A 16x32 layout: lanes 0-15 halves 0..7 = K k0..k0+7,
    // halves 8..15 = K k0+16..k0+23; lanes 16-31 use k0+8 base.
    v16bf afrag[4];
    const float* xr = x + (size_t)(m0 + lrow) * DDIM;
    #pragma unroll
    for (int c = 0; c < 4; ++c) {
        const float* s = xr + c * 32 + hs * 8;
        #pragma unroll
        for (int e = 0; e < 8; ++e) afrag[c][e]     = f2bf(s[e]);
        #pragma unroll
        for (int e = 0; e < 8; ++e) afrag[c][8 + e] = f2bf(s[16 + e]);
    }

    // Running argmax: packed f32 scores, column index in low 10 bits.
    float bestP[8];
    #pragma unroll
    for (int j = 0; j < 8; ++j) bestP[j] = -3.402823e38f;

#define LOADB(dst, t)                                                         \
    do {                                                                      \
        const __bf16* er_ = semb + (size_t)((t) * 16 + lrow) * LROW;          \
        _Pragma("unroll")                                                     \
        for (int c_ = 0; c_ < 4; ++c_) {                                      \
            v8bf lo_ = *(const v8bf*)(er_ + c_ * 32 + hs * 8);                \
            v8bf hi_ = *(const v8bf*)(er_ + c_ * 32 + hs * 8 + 16);           \
            dst[c_] = __builtin_shufflevector(                                \
                lo_, hi_, 0,1,2,3,4,5,6,7,8,9,10,11,12,13,14,15);             \
        }                                                                     \
    } while (0)

#define COMPUTE(bsrc, t, bc)                                                  \
    do {                                                                      \
        v8f acc_;                                                             \
        _Pragma("unroll")                                                     \
        for (int j_ = 0; j_ < 8; ++j_) acc_[j_] = (bc); /* C = -0.5||e||^2 */ \
        _Pragma("unroll")                                                     \
        for (int c_ = 0; c_ < 4; ++c_)                                        \
            acc_ = __builtin_amdgcn_wmma_f32_16x16x32_bf16(                   \
                false, afrag[c_], false, bsrc[c_], (short)0, acc_,            \
                false, false);                                                \
        const unsigned col_ = (unsigned)((t) * 16 + lrow);                    \
        _Pragma("unroll")                                                     \
        for (int j_ = 0; j_ < 8; ++j_) {                                      \
            float sc_ = acc_[j_];                                             \
            asm("" : "+v"(sc_));   /* keep 8 distinct element reads */        \
            const unsigned pb_ =                                              \
                (__builtin_bit_cast(unsigned, sc_) & IDXMASK) | col_;         \
            bestP[j_] = __builtin_fmaxf(bestP[j_],                            \
                                        __builtin_bit_cast(float, pb_));      \
        }                                                                     \
    } while (0)

    // ---- sweep all 64 code tiles; B frags + bias software-pipelined -------
    v16bf bping[4], bpong[4];
    float bc0 = sbias[lrow];                 // bias for tile 0
    LOADB(bping, 0);
    for (int t = 0; t < NTILES; t += 2) {
        const float bc1 = sbias[(t + 1) * 16 + lrow];   // issue before B clause
        LOADB(bpong, t + 1);
        COMPUTE(bping, t, bc0);
        float bcn = 0.0f;
        if (t + 2 < NTILES) {
            bcn = sbias[(t + 2) * 16 + lrow];
            LOADB(bping, t + 2);
        }
        COMPUTE(bpong, t + 1, bc1);
        bc0 = bcn;
    }
#undef LOADB
#undef COMPUTE

    // ---- per-row argmax across the 16 columns in each half-wave group -----
    // C/D layout: VGPR j, lanes 0-15 -> row m0+j; lanes 16-31 -> row m0+8+j.
    #pragma unroll
    for (int j = 0; j < 8; ++j) {
        float p = bestP[j];
        #pragma unroll
        for (int m = 8; m >= 1; m >>= 1)
            p = __builtin_fmaxf(p, __shfl_xor(p, m));
        const int bi  = (int)(__builtin_bit_cast(unsigned, p) & 0x3FFu);
        const int row = m0 + hs * 8 + j;
        if (lrow == 0) idxf[row] = (float)bi;
        // cooperative gather of the winning f32 codebook row (128 floats)
        const float* e = embed + (size_t)bi * DDIM + lrow * 8;
        const float4 q0 = *(const float4*)(e);
        const float4 q1 = *(const float4*)(e + 4);
        float* qo = quant + (size_t)row * DDIM + lrow * 8;
        *(float4*)(qo)     = q0;
        *(float4*)(qo + 4) = q1;
    }
}

extern "C" void kernel_launch(void* const* d_in, const int* in_sizes, int n_in,
                              void* d_out, int out_size, void* d_ws, size_t ws_size,
                              hipStream_t stream) {
    const float* x     = (const float*)d_in[0];
    const float* embed = (const float*)d_in[1];
    const int N = in_sizes[0] / DDIM;   // 65536 rows
    const int K = in_sizes[1] / DDIM;   // 1024 codes

    __bf16* ebf   = (__bf16*)d_ws;
    float*  nbias = (float*)((char*)d_ws + (size_t)K * DDIM * sizeof(unsigned short));
    float*  quant = (float*)d_out;
    float*  idxf  = quant + (size_t)N * DDIM;

    vq_prep_kernel<<<K, DDIM, 0, stream>>>(embed, ebf, nbias);
    vq_main_kernel<<<N / 128, 256, 0, stream>>>(x, embed, ebf, nbias, quant, idxf);
}
